// LSAAttention_44409961840851
// MI455X (gfx1250) — compile-verified, matrix operand order checked by reading, not run on hardware
//
#include <hip/hip_runtime.h>
#include <hip/hip_bf16.h>

// ---------------- problem constants ----------------
#define B_   2
#define T_   2048
#define DM_  1024
#define H_   16
#define DH_  64
#define DKV_ 256
#define DS_  64
#define W_   256
#define NT_  (B_ * T_)   // 4096 tokens

// ---------------- bf16 / WMMA plumbing ----------------
typedef __attribute__((ext_vector_type(16))) __bf16 v16bf;
typedef __attribute__((ext_vector_type(8)))  float  v8f;

union BF16x16 { unsigned short u[16]; uint4 q[2]; v16bf v; };

__device__ inline unsigned short f2bf(float f) {
  unsigned int u = __float_as_uint(f);
  u += 0x7FFFu + ((u >> 16) & 1u);       // round-to-nearest-even
  return (unsigned short)(u >> 16);
}
__device__ inline float bf2f(unsigned short u) {
  return __uint_as_float(((unsigned int)u) << 16);
}

// A-fragment (16x32 bf16): row M = lane%16; lanes 0-15 hold K 0..7/16..23,
// lanes 16-31 hold K 8..15/24..31. K is contiguous per lane -> 2x ds_load_b128.
// s is [row][k] row-major with `stride` (multiple of 8).
__device__ inline v16bf load_frag_a_vec(const unsigned short* s, int stride, int lane) {
  int row = lane & 15;
  int kb  = (lane >> 4) * 8;
  BF16x16 f;
  f.q[0] = *(const uint4*)(s + row * stride + kb);
  f.q[1] = *(const uint4*)(s + row * stride + 16 + kb);
  return f.v;
}
// B-fragment (32x16 bf16, K x N) from an N-major (transposed) tile:
// entry (k,n) at s[n*stride + k]; per-lane K contiguous -> 2x ds_load_b128.
__device__ inline v16bf load_frag_bt(const unsigned short* s, int stride, int lane, int colOff) {
  int col = colOff + (lane & 15);
  int kb  = (lane >> 4) * 8;
  BF16x16 f;
  f.q[0] = *(const uint4*)(s + col * stride + kb);
  f.q[1] = *(const uint4*)(s + col * stride + 16 + kb);
  return f.v;
}

// ---------------- generic bf16 WMMA GEMM ----------------
// C[M,N] = act(A[M,K] @ B[K,N] + bias).  Block tile 128x64, 8 waves stacked on
// M, each wave computes 16x64 (4 WMMAs per K-step, B-frags shared via LDS).
// Requires M%128==0, N%64==0, K%32==0.
#define BM 128
#define BN 64
#define BK 32
__global__ __launch_bounds__(256)
void gemm_bf16_kernel(const unsigned short* __restrict__ A,
                      const unsigned short* __restrict__ Bm,
                      const float* __restrict__ bias,
                      float* __restrict__ C,
                      int M, int N, int K, int act) {
  __shared__ __align__(16) unsigned short sA[BM * BK];   // [m][k]
  __shared__ __align__(16) unsigned short sBT[BN * BK];  // [n][k] transposed
  int tid  = threadIdx.x;
  int lane = tid & 31;
  int wave = tid >> 5;                  // 0..7 -> M sub-tile
  int m0 = blockIdx.y * BM;
  int n0 = blockIdx.x * BN;

  v8f acc[4] = {};
  for (int k0 = 0; k0 < K; k0 += BK) {
    // stage A: 128 rows x 32 k = 512 uint4, 2 per thread (b128 both sides)
    for (int i = tid; i < 512; i += 256) {
      int r = i >> 2, c = (i & 3) * 8;
      *(uint4*)(sA + r * BK + c) =
          *(const uint4*)(A + (size_t)(m0 + r) * K + k0 + c);
    }
    // stage B transposed: 32 k-rows x 64 n; vector global read, scatter to LDS
    {
      int r = tid >> 3, cb = (tid & 7) * 8;
      uint4 t = *(const uint4*)(Bm + (size_t)(k0 + r) * N + n0 + cb);
      const unsigned short* ts = (const unsigned short*)&t;
#pragma unroll
      for (int u = 0; u < 8; ++u) sBT[(cb + u) * BK + r] = ts[u];
    }
    // prefetch next K tile into cache while this one computes
    if (k0 + BK < K) {
      if (tid < BM)
        __builtin_prefetch(A + (size_t)(m0 + tid) * K + k0 + BK, 0, 1);
      else if (tid < BM + BK)
        __builtin_prefetch(Bm + (size_t)(k0 + BK + (tid - BM)) * N + n0, 0, 1);
    }
    __syncthreads();
    v16bf a = load_frag_a_vec(sA + wave * 16 * BK, BK, lane);
#pragma unroll
    for (int nt = 0; nt < 4; ++nt) {
      v16bf b = load_frag_bt(sBT + nt * 16 * BK, BK, lane, 0);
      acc[nt] = __builtin_amdgcn_wmma_f32_16x16x32_bf16(false, a, false, b,
                                                        (short)0, acc[nt],
                                                        false, false);
    }
    __syncthreads();
  }
  int nl = lane & 15;
#pragma unroll
  for (int nt = 0; nt < 4; ++nt) {
    union { v8f v; float f[8]; } res; res.v = acc[nt];
    int n = n0 + nt * 16 + nl;
    float bv = bias ? bias[n] : 0.f;     // bias has length N
#pragma unroll
    for (int r = 0; r < 8; ++r) {
      int m = m0 + wave * 16 + r + 8 * (lane >> 4);
      float v = res.f[r] + bv;
      if (act == 1) v = 1.f / (1.f + __expf(-v));   // sigmoid
      else if (act == 2) v = tanhf(v);              // tanh
      C[(size_t)m * N + n] = v;
    }
  }
}

// ---------------- f32 -> bf16 convert ----------------
__global__ void convert_f32_bf16(const float* __restrict__ in,
                                 unsigned short* __restrict__ outp, int n) {
  for (int i = blockIdx.x * blockDim.x + threadIdx.x; i < n;
       i += gridDim.x * blockDim.x)
    outp[i] = f2bf(in[i]);
}

// ---------------- RoPE (q,k) + convert to bf16 ----------------
__global__ void rope_kernel(const float* __restrict__ q, const float* __restrict__ k,
                            const float* __restrict__ cs, const float* __restrict__ sn,
                            unsigned short* __restrict__ qb,
                            unsigned short* __restrict__ kbo, int total) {
  int idx = blockIdx.x * blockDim.x + threadIdx.x;
  if (idx >= total) return;
  int d   = idx & 31;
  int h   = (idx >> 5) & (H_ - 1);
  int tok = idx >> 9;
  int t   = tok & (T_ - 1);
  float c = cs[t * 32 + d], s = sn[t * 32 + d];
  size_t base = (size_t)tok * DM_ + h * DH_;
  float x1 = q[base + d], x2 = q[base + 32 + d];
  qb[base + d]      = f2bf(x1 * c - x2 * s);
  qb[base + 32 + d] = f2bf(x2 * c + x1 * s);
  x1 = k[base + d]; x2 = k[base + 32 + d];
  kbo[base + d]      = f2bf(x1 * c - x2 * s);
  kbo[base + 32 + d] = f2bf(x2 * c + x1 * s);
}

// ---------------- serial gate scan: s = a*s + b*u ----------------
__global__ void scan_kernel(const float* __restrict__ Ag, const float* __restrict__ Bg,
                            const float* __restrict__ si, float* __restrict__ st) {
  int b = blockIdx.x, d = threadIdx.x;
  float s = 0.f;
  for (int t = 0; t < T_; ++t) {
    size_t i = ((size_t)(b * T_ + t)) * DS_ + d;
    s = Ag[i] * s + Bg[i] * si[i];
    st[i] = s;
  }
}

// ---------------- sliding-window attention + state slot ----------------
// One wave handles 16 query rows of one (b,h). Window W=256 causal =>
// keys in [t0-256, t0+15] = 17 tiles of 16.
#define SW_ 288   // score-row stride (17*16 = 272, padded)
__global__ __launch_bounds__(32)
void attn_kernel(const unsigned short* __restrict__ qb,
                 const unsigned short* __restrict__ kb,
                 const unsigned short* __restrict__ vb,
                 const float* __restrict__ k_state,
                 const float* __restrict__ v_state,
                 float* __restrict__ outp) {
  __shared__ __align__(16) unsigned short sQ[16 * 64];   // [row][feat]
  __shared__ __align__(16) unsigned short sK[16 * 64];   // [key][feat]
  __shared__ __align__(16) unsigned short sVT[64 * 32];  // [dh][key] transposed
  __shared__ float sS[16 * SW_];
  __shared__ float sRed[32];
  __shared__ float sDen[16];
  __shared__ float sPst[16];

  int lane = threadIdx.x;
  int t0 = blockIdx.x * 16;
  int h  = blockIdx.y;
  int b  = blockIdx.z;
  const size_t tokBase = (size_t)b * T_;
  const float scale = 0.125f;   // 1/sqrt(64)

  // stage Q: 16 rows x 64 feat = 128 uint4, vector both sides
  for (int i = lane; i < 128; i += 32) {
    int r = i >> 3, c = (i & 7) * 8;
    *(uint4*)(sQ + r * 64 + c) =
        *(const uint4*)(qb + (tokBase + t0 + r) * DM_ + h * DH_ + c);
  }
  for (int i = lane; i < 16 * SW_; i += 32) sS[i] = -1e30f;
  __syncthreads();

  v16bf aQ0 = load_frag_a_vec(sQ, 64, lane);        // features 0..31
  v16bf aQ1 = load_frag_a_vec(sQ + 32, 64, lane);   // features 32..63

  // ---- scores: Q @ K^T over 17 key tiles ----
  for (int kt = 0; kt < 17; ++kt) {
    int ks = t0 - 256 + kt * 16;   // tile-aligned: all >=0 or all <0
    if (ks < 0) continue;
    for (int i = lane; i < 128; i += 32) {
      int r = i >> 3, c = (i & 7) * 8;
      *(uint4*)(sK + r * 64 + c) =
          *(const uint4*)(kb + (tokBase + ks + r) * DM_ + h * DH_ + c);
    }
    __syncthreads();
    // B operand entry (k=feature, n=key) = sK[key][feature] (N-major)
    v16bf b0 = load_frag_bt(sK, 64, lane, 0);        // feat 0..31
    v16bf b1 = load_frag_bt(sK + 32, 64, lane, 0);   // feat 32..63
    v8f s0 = {};
    s0 = __builtin_amdgcn_wmma_f32_16x16x32_bf16(false, aQ0, false, b0,
                                                 (short)0, s0, false, false);
    s0 = __builtin_amdgcn_wmma_f32_16x16x32_bf16(false, aQ1, false, b1,
                                                 (short)0, s0, false, false);
    union { v8f v; float f[8]; } sc; sc.v = s0;
    int n = lane & 15;
#pragma unroll
    for (int r = 0; r < 8; ++r) {
      int m = r + 8 * (lane >> 4);
      int t = t0 + m;
      int j = ks + n;
      bool ok = (j <= t) && (j >= t - (W_ - 1));
      sS[m * SW_ + kt * 16 + n] = ok ? sc.f[r] * scale : -1e30f;
    }
    __syncthreads();
  }

  // ---- state score: dot(q[t], k_state[t]) ----
  {
    int row = lane & 15, half = lane >> 4;
    float p = 0.f;
    const float* ksr = k_state + (tokBase + t0 + row) * DM_ + h * DH_ + half * 32;
    const unsigned short* qr = sQ + row * 64 + half * 32;
    for (int d2 = 0; d2 < 32; ++d2) p += bf2f(qr[d2]) * ksr[d2];
    sRed[lane] = p;
  }
  __syncthreads();

  // ---- softmax over (272 local + state) per row ----
  if (lane < 16) {
    float st = (sRed[lane] + sRed[lane + 16]) * scale;
    float mx = st;
    for (int c = 0; c < 272; ++c) mx = fmaxf(mx, sS[lane * SW_ + c]);
    float den = 0.f;
    for (int c = 0; c < 272; ++c) {
      float e = __expf(sS[lane * SW_ + c] - mx);
      sS[lane * SW_ + c] = e;
      den += e;
    }
    for (int c = 272; c < SW_; ++c) sS[lane * SW_ + c] = 0.f;
    float pst = __expf(st - mx);
    sDen[lane] = den + pst;
    sPst[lane] = pst;
  }
  __syncthreads();

  // ---- out = P @ V over 9 chunks of 32 keys, 4 DH tiles ----
  v8f accO[4] = {};
  for (int kc = 0; kc < 9; ++kc) {
    // stage V transposed: vector global read, scatter to [dh][key]
    for (int i = lane; i < 256; i += 32) {
      int key = i >> 3, db = (i & 7) * 8;
      int j = t0 - 256 + kc * 32 + key;
      if (j >= 0 && j < T_) {
        uint4 t = *(const uint4*)(vb + (tokBase + j) * DM_ + h * DH_ + db);
        const unsigned short* ts = (const unsigned short*)&t;
#pragma unroll
        for (int u = 0; u < 8; ++u) sVT[(db + u) * 32 + key] = ts[u];
      } else {
#pragma unroll
        for (int u = 0; u < 8; ++u) sVT[(db + u) * 32 + key] = 0;
      }
    }
    __syncthreads();
    BF16x16 ap;
    {
      int row = lane & 15; int kbx = (lane >> 4) * 8;
#pragma unroll
      for (int i2 = 0; i2 < 8; ++i2) {
        ap.u[i2]     = f2bf(sS[row * SW_ + kc * 32 + kbx + i2]);
        ap.u[8 + i2] = f2bf(sS[row * SW_ + kc * 32 + 16 + kbx + i2]);
      }
    }
#pragma unroll
    for (int nt = 0; nt < 4; ++nt) {
      v16bf bv = load_frag_bt(sVT, 32, lane, nt * 16);  // (k=key, n=dh)
      accO[nt] = __builtin_amdgcn_wmma_f32_16x16x32_bf16(false, ap.v, false, bv,
                                                         (short)0, accO[nt],
                                                         false, false);
    }
    __syncthreads();
  }

  // ---- epilogue: add state value, normalize ----
#pragma unroll
  for (int nt = 0; nt < 4; ++nt) {
    union { v8f v; float f[8]; } o; o.v = accO[nt];
    int n = nt * 16 + (lane & 15);
#pragma unroll
    for (int r = 0; r < 8; ++r) {
      int m = r + 8 * (lane >> 4);
      int t = t0 + m;
      float vs = v_state[(tokBase + t) * DM_ + h * DH_ + n];
      outp[(tokBase + t) * DM_ + h * DH_ + n] =
          (o.f[r] + sPst[m] * vs) / sDen[m];
    }
  }
}

// ---------------- host orchestration ----------------
extern "C" void kernel_launch(void* const* d_in, const int* in_sizes, int n_in,
                              void* d_out, int out_size, void* d_ws, size_t ws_size,
                              hipStream_t stream) {
  (void)in_sizes; (void)n_in; (void)out_size; (void)ws_size;
  const float* x    = (const float*)d_in[0];
  const float* rc   = (const float*)d_in[1];
  const float* rs   = (const float*)d_in[2];
  const float* Wq   = (const float*)d_in[3];
  const float* Wkvd = (const float*)d_in[4];
  const float* Wku  = (const float*)d_in[5];
  const float* Wvu  = (const float*)d_in[6];
  const float* WA   = (const float*)d_in[7];
  const float* bA   = (const float*)d_in[8];
  const float* WB   = (const float*)d_in[9];
  const float* bB   = (const float*)d_in[10];
  const float* Wins = (const float*)d_in[11];
  const float* Wsk  = (const float*)d_in[12];
  const float* Wsv  = (const float*)d_in[13];
  const float* Wo   = (const float*)d_in[14];

  char* ws = (char*)d_ws;
  size_t off = 0;
  auto alloc = [&](size_t bytes) -> void* {
    void* p = ws + off;
    off = (off + bytes + 255) & ~(size_t)255;
    return p;
  };

  // bf16 buffers
  unsigned short* xb       = (unsigned short*)alloc((size_t)NT_ * DM_ * 2);
  unsigned short* Wq_b     = (unsigned short*)alloc((size_t)DM_ * DM_ * 2);
  unsigned short* Wkvd_b   = (unsigned short*)alloc((size_t)DM_ * DKV_ * 2);
  unsigned short* WA_b     = (unsigned short*)alloc((size_t)DM_ * DS_ * 2);
  unsigned short* WB_b     = (unsigned short*)alloc((size_t)DM_ * DS_ * 2);
  unsigned short* Wku_b    = (unsigned short*)alloc((size_t)DKV_ * DM_ * 2);
  unsigned short* Wvu_b    = (unsigned short*)alloc((size_t)DKV_ * DM_ * 2);
  unsigned short* Wins_b   = (unsigned short*)alloc((size_t)DKV_ * DS_ * 2);
  unsigned short* Wsk_b    = (unsigned short*)alloc((size_t)DS_ * DM_ * 2);
  unsigned short* Wsv_b    = (unsigned short*)alloc((size_t)DS_ * DM_ * 2);
  unsigned short* Wo_b     = (unsigned short*)alloc((size_t)DM_ * DM_ * 2);
  unsigned short* c_kvb    = (unsigned short*)alloc((size_t)NT_ * DKV_ * 2);
  unsigned short* qb       = (unsigned short*)alloc((size_t)NT_ * DM_ * 2);
  unsigned short* k_localb = (unsigned short*)alloc((size_t)NT_ * DM_ * 2);
  unsigned short* v_localb = (unsigned short*)alloc((size_t)NT_ * DM_ * 2);
  unsigned short* statesb  = (unsigned short*)alloc((size_t)NT_ * DS_ * 2);
  unsigned short* attnb    = (unsigned short*)alloc((size_t)NT_ * DM_ * 2);

  // f32 buffers (bigA/bigB/bigC reused across phases)
  float* bigA   = (float*)alloc((size_t)NT_ * DM_ * 4);  // q_f32 -> attn_out
  float* bigB   = (float*)alloc((size_t)NT_ * DM_ * 4);  // k_local_f32 -> k_state
  float* bigC   = (float*)alloc((size_t)NT_ * DM_ * 4);  // v_local_f32 -> v_state
  float* c_kv   = (float*)alloc((size_t)NT_ * DKV_ * 4);
  float* Ag     = (float*)alloc((size_t)NT_ * DS_ * 4);
  float* Bg     = (float*)alloc((size_t)NT_ * DS_ * 4);
  float* si     = (float*)alloc((size_t)NT_ * DS_ * 4);
  float* states = (float*)alloc((size_t)NT_ * DS_ * 4);

  auto conv = [&](const float* in, unsigned short* outp, int n) {
    int blocks = (n + 255) / 256; if (blocks > 8192) blocks = 8192;
    convert_f32_bf16<<<blocks, 256, 0, stream>>>(in, outp, n);
  };
  auto gemm = [&](const unsigned short* A, const unsigned short* Bmat,
                  const float* bias, float* C, int M, int N, int K, int act) {
    dim3 g(N / BN, M / BM);
    gemm_bf16_kernel<<<g, 256, 0, stream>>>(A, Bmat, bias, C, M, N, K, act);
  };

  // 1. bf16 conversions
  conv(x,    xb,     NT_ * DM_);
  conv(Wq,   Wq_b,   DM_ * DM_);
  conv(Wkvd, Wkvd_b, DM_ * DKV_);
  conv(WA,   WA_b,   DM_ * DS_);
  conv(WB,   WB_b,   DM_ * DS_);
  conv(Wku,  Wku_b,  DKV_ * DM_);
  conv(Wvu,  Wvu_b,  DKV_ * DM_);
  conv(Wins, Wins_b, DKV_ * DS_);
  conv(Wsk,  Wsk_b,  DS_ * DM_);
  conv(Wsv,  Wsv_b,  DS_ * DM_);
  conv(Wo,   Wo_b,   DM_ * DM_);

  // 2. projections from x
  gemm(xb, Wq_b,   nullptr, bigA, NT_, DM_,  DM_, 0);  // q (pre-RoPE)
  gemm(xb, Wkvd_b, nullptr, c_kv, NT_, DKV_, DM_, 0);  // c_kv
  gemm(xb, WA_b,   bA,      Ag,   NT_, DS_,  DM_, 1);  // sigmoid gate
  gemm(xb, WB_b,   bB,      Bg,   NT_, DS_,  DM_, 2);  // tanh gate

  // 3. up-projections from c_kv
  conv(c_kv, c_kvb, NT_ * DKV_);
  gemm(c_kvb, Wku_b,  nullptr, bigB, NT_, DM_, DKV_, 0);  // k_local (pre-RoPE)
  gemm(c_kvb, Wvu_b,  nullptr, bigC, NT_, DM_, DKV_, 0);  // v_local
  gemm(c_kvb, Wins_b, nullptr, si,   NT_, DS_, DKV_, 0);  // state input

  // 4. RoPE(q,k) -> bf16, v_local -> bf16
  {
    int total = NT_ * H_ * 32;
    rope_kernel<<<(total + 255) / 256, 256, 0, stream>>>(bigA, bigB, rc, rs,
                                                         qb, k_localb, total);
  }
  conv(bigC, v_localb, NT_ * DM_);

  // 5. gate scan + state projections (bigB/bigC reused as k_state/v_state)
  scan_kernel<<<B_, DS_, 0, stream>>>(Ag, Bg, si, states);
  conv(states, statesb, NT_ * DS_);
  gemm(statesb, Wsk_b, nullptr, bigB, NT_, DM_, DS_, 0);  // k_state
  gemm(statesb, Wsv_b, nullptr, bigC, NT_, DM_, DS_, 0);  // v_state

  // 6. attention (bigA reused as attn output)
  {
    dim3 g(T_ / 16, H_, B_);
    attn_kernel<<<g, 32, 0, stream>>>(qb, k_localb, v_localb, bigB, bigC, bigA);
  }

  // 7. output projection
  conv(bigA, attnb, NT_ * DM_);
  gemm(attnb, Wo_b, nullptr, (float*)d_out, NT_, DM_, DM_, 0);
}
